// EncoderBlock_39754217292479
// MI455X (gfx1250) — matrix operand order repaired and checked by reading, not compiled, not run on hardware
//
#include <hip/hip_runtime.h>

// ---------------------------------------------------------------------------
// EncoderBlock for MI455X (gfx1250): B=2,S=2048,D=1024,H=16,DK=64,DFF=4096
// GEMMs via v_wmma_f32_16x16x32_f16; weight tiles double-buffered into LDS by
// the Tensor Data Mover (tensor_load_to_lds); attention is flash-style.
// ---------------------------------------------------------------------------

typedef __attribute__((ext_vector_type(16))) _Float16 v16h;
typedef __attribute__((ext_vector_type(8)))  _Float16 v8h;
typedef __attribute__((ext_vector_type(4)))  _Float16 v4h;
typedef __attribute__((ext_vector_type(8)))  float    v8f;
typedef unsigned int u32;
typedef __attribute__((ext_vector_type(4))) u32 v4u;
typedef __attribute__((ext_vector_type(8))) int v8i;
typedef __attribute__((ext_vector_type(4))) int v4i;

// TDM availability (device pass only; host pass parses the fallback path)
#if defined(__HIP_DEVICE_COMPILE__) && defined(__has_builtin)
#if __has_builtin(__builtin_amdgcn_tensor_load_to_lds) && \
    __has_builtin(__builtin_amdgcn_s_wait_tensorcnt)
#define USE_TDM 1
#endif
#endif
#ifndef USE_TDM
#define USE_TDM 0
#endif

#if defined(__HIP_DEVICE_COMPILE__)
#if USE_TDM
#pragma message("CDNA5: GEMM uses tensor_load_to_lds (TDM) double-buffered path")
#else
#pragma message("CDNA5: GEMM uses cooperative-load fallback path (no TDM builtin)")
#endif
#endif

constexpr int B_   = 2;
constexpr int S_   = 2048;
constexpr int D_   = 1024;
constexpr int H_   = 16;
constexpr int DFF_ = 4096;
constexpr int DK_  = D_ / H_;   // 64
constexpr int M_   = B_ * S_;   // 4096
constexpr float EPS_ = 1e-6f;

__device__ __forceinline__ v8f wmma16(v16h a, v16h b, v8f c) {
  // D = A(16x32 f16) x B(32x16 f16) + C(16x16 f32)
  return __builtin_amdgcn_wmma_f32_16x16x32_f16(false, a, false, b, (short)0, c,
                                                false, false);
}

// A-fragment (16x32 f16) load from row-major [., lda] matrix.
// Layout: lane L holds row m=L&15; halves 0..15 -> K = g*16+i,
// with g = L>>4.
__device__ __forceinline__ v16h load_a_frag(const _Float16* base, int lda,
                                            int row0, int k0, int lm, int lg) {
  const _Float16* p = base + (size_t)(row0 + lm) * lda + k0 + lg * 16;
  v8h lo = *(const v8h*)(p);
  v8h hi = *(const v8h*)(p + 8);
  v16h a;
#pragma unroll
  for (int i = 0; i < 8; ++i) { a[i] = lo[i]; a[i + 8] = hi[i]; }
  return a;
}

// ---------------------------------------------------------------------------
// f32 -> f16 conversion (weights), vectorized x4
// ---------------------------------------------------------------------------
__global__ void __launch_bounds__(256) cvt_f16_kernel(
    const float* __restrict__ src, _Float16* __restrict__ dst, int n4) {
  int i = blockIdx.x * blockDim.x + threadIdx.x;
  if (i < n4) {
    float4 v = ((const float4*)src)[i];
    v4h h;
    h[0] = (_Float16)v.x; h[1] = (_Float16)v.y;
    h[2] = (_Float16)v.z; h[3] = (_Float16)v.w;
    ((v4h*)dst)[i] = h;
  }
}

// ---------------------------------------------------------------------------
// LayerNorm (torch std: ddof=1, divide by (std+eps)); one row per block.
// D=1024, 256 threads x 4 floats. Output f16.
// ---------------------------------------------------------------------------
__global__ void __launch_bounds__(256) ln_kernel(
    const float* __restrict__ x, const float* __restrict__ alpha,
    const float* __restrict__ beta, _Float16* __restrict__ out) {
  const int row = blockIdx.x;
  const int tid = threadIdx.x;
  const int wid = tid >> 5, lane = tid & 31;
  __shared__ float sb[8];

  const float4 v = ((const float4*)(x + (size_t)row * D_))[tid];
  float s = v.x + v.y + v.z + v.w;
#pragma unroll
  for (int o = 16; o; o >>= 1) s += __shfl_xor(s, o, 32);
  if (lane == 0) sb[wid] = s;
  __syncthreads();
  float tot = 0.f;
#pragma unroll
  for (int i = 0; i < 8; ++i) tot += sb[i];
  const float mean = tot * (1.0f / D_);

  const float c0 = v.x - mean, c1 = v.y - mean, c2 = v.z - mean, c3 = v.w - mean;
  float sq = c0 * c0 + c1 * c1 + c2 * c2 + c3 * c3;
#pragma unroll
  for (int o = 16; o; o >>= 1) sq += __shfl_xor(sq, o, 32);
  __syncthreads();
  if (lane == 0) sb[wid] = sq;
  __syncthreads();
  float tq = 0.f;
#pragma unroll
  for (int i = 0; i < 8; ++i) tq += sb[i];
  const float stdv = __builtin_sqrtf(tq * (1.0f / (D_ - 1)));
  const float inv  = alpha[0] / (stdv + EPS_);
  const float bet  = beta[0];

  v4h h;
  h[0] = (_Float16)(c0 * inv + bet);
  h[1] = (_Float16)(c1 * inv + bet);
  h[2] = (_Float16)(c2 * inv + bet);
  h[3] = (_Float16)(c3 * inv + bet);
  ((v4h*)(out + (size_t)row * D_))[tid] = h;
}

// ---------------------------------------------------------------------------
// WMMA GEMM: C[M x N] = A_f16[M x K] * W_f16[N x K]^T (+bias, epilogue EPI)
//   EPI 0: f16 store to head-split [B,H,S,DK]       (Q, K projections)
//   EPI 1: f16 store transposed    [B,H,DK,S]       (V projection)
//   EPI 2: f32 store row-major, + bias + residual   (Wo + x, FFN2 + h)
//   EPI 3: f16 store row-major, relu(.+bias)        (FFN1)
// Block: 256 thr = 8 waves; block tile 256(M) x 64(N); per-wave tile 32x64
// (2 M-tiles x 4 N-tiles = 8 WMMAs per k-step). The 64x32 f16 weight tile is
// DMA'd into LDS by the Tensor Data Mover, double-buffered: wave 0 issues the
// DMA for tile i+1, waits tensorcnt<=1 (tile i resident, TDM is in-order),
// then the block barrier releases compute on buffer i&1.
// ---------------------------------------------------------------------------
template <int EPI>
__global__ void __launch_bounds__(256, 1) gemm_kernel(
    const _Float16* __restrict__ A, const _Float16* __restrict__ W,
    const float* __restrict__ bias, const float* __restrict__ resid,
    void* __restrict__ outv, int N, int K) {
  __shared__ __align__(64) _Float16 sh[2][64 * 32];
  const int tid  = threadIdx.x;
  const int wid  = tid >> 5;
  const int lane = tid & 31;
  const int lm = lane & 15, lg = lane >> 4;
  const int row0 = blockIdx.y * 256 + wid * 32;
  const int n0   = blockIdx.x * 64;

  v8f acc[2][4] = {};

#if USE_TDM
  const u32 lds_base = (u32)(uintptr_t)(void*)&sh[0][0];
  const unsigned long long wtile0 =
      (unsigned long long)(uintptr_t)(W + (size_t)n0 * K);
  // Issue a 64x32 f16 tile DMA (row stride K elements) into buffer `buf`.
  auto tdm_issue = [&](int kk, int buf) {
    const unsigned long long ga = wtile0 + (unsigned long long)kk * 2ull;
    v4u g0;
    g0[0] = 1u;                                  // count=1 (valid), user mode
    g0[1] = lds_base + (u32)buf * (64u * 32u * 2u);  // lds_addr
    g0[2] = (u32)ga;                             // global_addr[31:0]
    g0[3] = (u32)(ga >> 32) | (2u << 30);        // global_addr[56:32] | type=2
    v8i g1;
    g1[0] = (int)(1u << 16);                     // data_size=1 -> 2 bytes
    g1[1] = (int)(((u32)K & 0xffffu) << 16);     // tensor_dim0[15:0]
    g1[2] = (int)((((u32)K >> 16) & 0xffffu) |   // tensor_dim0[31:16]
                  (((u32)N & 0xffffu) << 16));   // tensor_dim1[15:0]
    g1[3] = (int)((((u32)N >> 16) & 0xffffu) |   // tensor_dim1[31:16]
                  (32u << 16));                  // tile_dim0 = 32
    g1[4] = 64;                                  // tile_dim1=64, tile_dim2=0
    g1[5] = (int)(u32)K;                         // tensor_dim0_stride[31:0]
    g1[6] = 0;                                   // stride hi, dim1_stride lo
    g1[7] = 0;
    v4i gz = {0, 0, 0, 0};
#if __clang_major__ >= 23
    v8i gz8 = {0, 0, 0, 0, 0, 0, 0, 0};
    __builtin_amdgcn_tensor_load_to_lds(g0, g1, gz, gz, gz8, 0);
#else
    __builtin_amdgcn_tensor_load_to_lds(g0, g1, gz, gz, 0);
#endif
  };

  const int nsteps = K / 32;
  if (wid == 0) tdm_issue(0, 0);
  for (int i = 0; i < nsteps; ++i) {
    const int kk = i * 32;
    if (wid == 0) {
      if (i + 1 < nsteps) {
        tdm_issue(kk + 32, (i + 1) & 1);
        __builtin_amdgcn_s_wait_tensorcnt(1);  // tile i resident (in-order)
      } else {
        __builtin_amdgcn_s_wait_tensorcnt(0);
      }
    }
    __syncthreads();
    const _Float16* shp = sh[i & 1];
    const v16h a0 = load_a_frag(A, K, row0,      kk, lm, lg);
    const v16h a1 = load_a_frag(A, K, row0 + 16, kk, lm, lg);
#pragma unroll
    for (int nt = 0; nt < 4; ++nt) {
      // B-frag: b[i] = W[n][kk + g*16 + i] = shp[n*32 + g*16 + i]
      const v16h b = *(const v16h*)(shp + (nt * 16 + lm) * 32 + lg * 16);
      acc[0][nt] = wmma16(a0, b, acc[0][nt]);
      acc[1][nt] = wmma16(a1, b, acc[1][nt]);
    }
    __syncthreads();  // buffer i&1 consumed; safe to re-target at i+2
  }
#else
  const int nrow = tid >> 2;          // 0..63 (local n)
  const int kcol = (tid & 3) * 8;     // 0,8,16,24 (local k)
  const _Float16* wsrc = W + (size_t)(n0 + nrow) * K + kcol;
  for (int kk = 0; kk < K; kk += 32) {
    __syncthreads();
    *(v8h*)(&sh[0][0] + nrow * 32 + kcol) = *(const v8h*)(wsrc + kk);
    if (kk + 32 < K) __builtin_prefetch(wsrc + kk + 32, 0, 1);
    __syncthreads();
    const v16h a0 = load_a_frag(A, K, row0,      kk, lm, lg);
    const v16h a1 = load_a_frag(A, K, row0 + 16, kk, lm, lg);
#pragma unroll
    for (int nt = 0; nt < 4; ++nt) {
      const v16h b = *(const v16h*)(&sh[0][0] + (nt * 16 + lm) * 32 + lg * 16);
      acc[0][nt] = wmma16(a0, b, acc[0][nt]);
      acc[1][nt] = wmma16(a1, b, acc[1][nt]);
    }
  }
#endif

  // D-layout: lane holds column n (= lane&15 within tile), rows m = r + 8*lg.
#pragma unroll
  for (int mt = 0; mt < 2; ++mt) {
    const int rowm = row0 + mt * 16;
#pragma unroll
    for (int nt = 0; nt < 4; ++nt) {
      const int n = n0 + nt * 16 + lm;
      const float bn = bias[n];
      if (EPI == 0) {
        _Float16* dst = (_Float16*)outv;
        const int hh = n >> 6, dk = n & (DK_ - 1);
#pragma unroll
        for (int r = 0; r < 8; ++r) {
          const int m = rowm + r + 8 * lg;
          const int bb = m >> 11, s = m & (S_ - 1);
          dst[((size_t)(bb * H_ + hh) * S_ + s) * DK_ + dk] =
              (_Float16)(acc[mt][nt][r] + bn);
        }
      } else if (EPI == 1) {
        _Float16* dst = (_Float16*)outv;
        const int hh = n >> 6, dk = n & (DK_ - 1);
        const int m0 = rowm + 8 * lg;
        const int bb = m0 >> 11, sb0 = m0 & (S_ - 1);
        v8h pk;
#pragma unroll
        for (int r = 0; r < 8; ++r) pk[r] = (_Float16)(acc[mt][nt][r] + bn);
        *(v8h*)(dst + ((size_t)(bb * H_ + hh) * DK_ + dk) * S_ + sb0) = pk;
      } else if (EPI == 2) {
        float* dst = (float*)outv;
#pragma unroll
        for (int r = 0; r < 8; ++r) {
          const int m = rowm + r + 8 * lg;
          const size_t idx = (size_t)m * N + n;
          dst[idx] = acc[mt][nt][r] + bn + resid[idx];
        }
      } else {  // EPI == 3
        _Float16* dst = (_Float16*)outv;
#pragma unroll
        for (int r = 0; r < 8; ++r) {
          const int m = rowm + r + 8 * lg;
          const float v = acc[mt][nt][r] + bn;
          dst[(size_t)m * N + n] = (_Float16)(v > 0.f ? v : 0.f);
        }
      }
    }
  }
}

// ---------------------------------------------------------------------------
// Flash attention: one wave per (b,h,16-query tile). Q,K in [B,H,S,DK] f16,
// V transposed in [B,H,DK,S] f16. Scores computed TRANSPOSED as
// T(keys x queries) = K_tile(A) x Q^T(B) so exp(T) lands directly in the
// f16 A-fragment layout of P for the P x V WMMA (zero cross-lane shuffles).
// ---------------------------------------------------------------------------
__global__ void __launch_bounds__(128, 1) attn_kernel(
    const _Float16* __restrict__ Q, const _Float16* __restrict__ Kp,
    const _Float16* __restrict__ Vt, _Float16* __restrict__ O) {
  const int lane = threadIdx.x & 31;
  const int lm = lane & 15, lg = lane >> 4;
  const int gw = blockIdx.x * 4 + (threadIdx.x >> 5);
  const int qt = gw & (S_ / 16 - 1);  // 0..127
  const int bh = gw >> 7;             // 0..31 (b*H + h)
  const int q0 = qt * 16;

  const _Float16* Qb = Q  + (size_t)bh * S_ * DK_;
  const _Float16* Kb = Kp + (size_t)bh * S_ * DK_;
  const _Float16* Vb = Vt + (size_t)bh * DK_ * S_;

  // Q^T B-fragments (fixed for this wave): b[i] = Q[q0+lm][d0 + g*16 + i]
  const v16h bq0 = *(const v16h*)(Qb + (size_t)(q0 + lm) * DK_ + lg * 16);
  const v16h bq1 = *(const v16h*)(Qb + (size_t)(q0 + lm) * DK_ + 32 + lg * 16);

  v8f o[4] = {};
  float mrun = -__builtin_inff();
  float lrun = 0.f;
  const float sc = 0.125f;  // 1/sqrt(DK)

  for (int j = 0; j < S_; j += 32) {
    // K A-fragments: rows are keys
    const v16h a00 = load_a_frag(Kb, DK_, j,      0,  lm, lg);
    const v16h a01 = load_a_frag(Kb, DK_, j,      32, lm, lg);
    const v16h a10 = load_a_frag(Kb, DK_, j + 16, 0,  lm, lg);
    const v16h a11 = load_a_frag(Kb, DK_, j + 16, 32, lm, lg);
    v8f t0 = {}; t0 = wmma16(a00, bq0, t0); t0 = wmma16(a01, bq1, t0);
    v8f t1 = {}; t1 = wmma16(a10, bq0, t1); t1 = wmma16(a11, bq1, t1);

    // online softmax over keys (T layout: lane's column = query lm,
    // lane holds keys {8*lg..8*lg+7} of chunk0 (t0) and chunk1 (t1))
    float cm = -__builtin_inff();
#pragma unroll
    for (int r = 0; r < 8; ++r)
      cm = fmaxf(cm, fmaxf(t0[r], t1[r]));
    cm *= sc;
    cm = fmaxf(cm, __shfl_xor(cm, 16, 32));
    const float mnew = fmaxf(mrun, cm);
    const float corr = __expf(mrun - mnew);
    mrun = mnew;

    float csum = 0.f;
    v16h ap;
#pragma unroll
    for (int r = 0; r < 8; ++r) {
      const float p0 = __expf(t0[r] * sc - mnew);
      const float p1 = __expf(t1[r] * sc - mnew);
      csum += p0 + p1;
      ap[r]     = (_Float16)p0;   // A-frag halves 0..7  -> keys g*8+i
      ap[r + 8] = (_Float16)p1;   // A-frag halves 8..15 -> keys 16+g*8+i
    }
    csum += __shfl_xor(csum, 16, 32);
    lrun = lrun * corr + csum;

    // rescale O accumulators: row q = r + 8*lg needs corr from lane q
    float cq[8];
#pragma unroll
    for (int r = 0; r < 8; ++r) cq[r] = __shfl(corr, r + 8 * lg, 32);
#pragma unroll
    for (int nt = 0; nt < 4; ++nt)
#pragma unroll
      for (int r = 0; r < 8; ++r) o[nt][r] *= cq[r];

    // P(16q x 32k) x V(32k x 64d): V B-frags contiguous from transposed V
#pragma unroll
    for (int nt = 0; nt < 4; ++nt) {
      const _Float16* vp = Vb + (size_t)(nt * 16 + lm) * S_ + j + lg * 8;
      const v8h vlo = *(const v8h*)(vp);
      const v8h vhi = *(const v8h*)(vp + 16);
      const v16h bv = __builtin_shufflevector(vlo, vhi, 0, 1, 2, 3, 4, 5, 6, 7,
                                              8, 9, 10, 11, 12, 13, 14, 15);
      o[nt] = wmma16(ap, bv, o[nt]);
    }
  }

  float rq[8];
#pragma unroll
  for (int r = 0; r < 8; ++r) rq[r] = 1.0f / __shfl(lrun, r + 8 * lg, 32);

  const int bb = bh >> 4, hh = bh & (H_ - 1);
#pragma unroll
  for (int nt = 0; nt < 4; ++nt) {
    const int col = hh * DK_ + nt * 16 + lm;
#pragma unroll
    for (int r = 0; r < 8; ++r) {
      const int q = q0 + r + 8 * lg;
      O[(size_t)(bb * S_ + q) * D_ + col] = (_Float16)(o[nt][r] * rq[r]);
    }
  }
}

// ---------------------------------------------------------------------------
// Launch
// ---------------------------------------------------------------------------
extern "C" void kernel_launch(void* const* d_in, const int* in_sizes, int n_in,
                              void* d_out, int out_size, void* d_ws,
                              size_t ws_size, hipStream_t stream) {
  (void)in_sizes; (void)n_in; (void)out_size; (void)ws_size;
  const float* x     = (const float*)d_in[0];
  // d_in[1] = mask (unused by the reference math)
  const float* wq = (const float*)d_in[2];
  const float* bq = (const float*)d_in[3];
  const float* wk = (const float*)d_in[4];
  const float* bk = (const float*)d_in[5];
  const float* wv = (const float*)d_in[6];
  const float* bv = (const float*)d_in[7];
  const float* wo = (const float*)d_in[8];
  const float* bo = (const float*)d_in[9];
  const float* w1 = (const float*)d_in[10];
  const float* b1 = (const float*)d_in[11];
  const float* w2 = (const float*)d_in[12];
  const float* b2 = (const float*)d_in[13];
  const float* ln1_a = (const float*)d_in[14];
  const float* ln1_b = (const float*)d_in[15];
  const float* ln2_a = (const float*)d_in[16];
  const float* ln2_b = (const float*)d_in[17];
  float* out = (float*)d_out;

  // ---- workspace carve (≈120 MB) ----
  char* w = (char*)d_ws;
  size_t off = 0;
  auto carve = [&](size_t bytes) {
    void* p = w + off;
    off += (bytes + 255) & ~(size_t)255;
    return p;
  };
  _Float16* wq16 = (_Float16*)carve((size_t)D_ * D_ * 2);
  _Float16* wk16 = (_Float16*)carve((size_t)D_ * D_ * 2);
  _Float16* wv16 = (_Float16*)carve((size_t)D_ * D_ * 2);
  _Float16* wo16 = (_Float16*)carve((size_t)D_ * D_ * 2);
  _Float16* w116 = (_Float16*)carve((size_t)DFF_ * D_ * 2);
  _Float16* w216 = (_Float16*)carve((size_t)D_ * DFF_ * 2);
  _Float16* ln1o = (_Float16*)carve((size_t)M_ * D_ * 2);
  _Float16* q16  = (_Float16*)carve((size_t)M_ * D_ * 2);
  _Float16* k16  = (_Float16*)carve((size_t)M_ * D_ * 2);
  _Float16* vT16 = (_Float16*)carve((size_t)M_ * D_ * 2);
  _Float16* at16 = (_Float16*)carve((size_t)M_ * D_ * 2);
  float*    h32  = (float*)   carve((size_t)M_ * D_ * 4);
  _Float16* ln2o = (_Float16*)carve((size_t)M_ * D_ * 2);
  _Float16* mid  = (_Float16*)carve((size_t)M_ * DFF_ * 2);

  // ---- weight conversion f32 -> f16 ----
  {
    const int n4dd = D_ * D_ / 4;       // 262144
    const int n4df = DFF_ * D_ / 4;     // 1048576
    cvt_f16_kernel<<<n4dd / 256, 256, 0, stream>>>(wq, wq16, n4dd);
    cvt_f16_kernel<<<n4dd / 256, 256, 0, stream>>>(wk, wk16, n4dd);
    cvt_f16_kernel<<<n4dd / 256, 256, 0, stream>>>(wv, wv16, n4dd);
    cvt_f16_kernel<<<n4dd / 256, 256, 0, stream>>>(wo, wo16, n4dd);
    cvt_f16_kernel<<<n4df / 256, 256, 0, stream>>>(w1, w116, n4df);
    cvt_f16_kernel<<<n4df / 256, 256, 0, stream>>>(w2, w216, n4df);
  }

  // ---- LN1 ----
  ln_kernel<<<M_, 256, 0, stream>>>(x, ln1_a, ln1_b, ln1o);

  // ---- QKV projections ----
  const dim3 gDD(D_ / 64, M_ / 256);    // (16, 16)
  gemm_kernel<0><<<gDD, 256, 0, stream>>>(ln1o, wq16, bq, nullptr, q16, D_, D_);
  gemm_kernel<0><<<gDD, 256, 0, stream>>>(ln1o, wk16, bk, nullptr, k16, D_, D_);
  gemm_kernel<1><<<gDD, 256, 0, stream>>>(ln1o, wv16, bv, nullptr, vT16, D_, D_);

  // ---- attention (flash) ----
  attn_kernel<<<(B_ * H_ * (S_ / 16)) / 4, 128, 0, stream>>>(q16, k16, vT16,
                                                             at16);

  // ---- output projection + residual 1 -> h (f32) ----
  gemm_kernel<2><<<gDD, 256, 0, stream>>>(at16, wo16, bo, x, h32, D_, D_);

  // ---- LN2 ----
  ln_kernel<<<M_, 256, 0, stream>>>(h32, ln2_a, ln2_b, ln2o);

  // ---- FFN ----
  const dim3 gDF(DFF_ / 64, M_ / 256);  // (64, 16)
  gemm_kernel<3><<<gDF, 256, 0, stream>>>(ln2o, w116, b1, nullptr, mid, DFF_, D_);
  gemm_kernel<2><<<gDD, 256, 0, stream>>>(mid, w216, b2, h32, out, D_, DFF_);
}
